// Codebook_17875653886031
// MI455X (gfx1250) — matrix operand order repaired and checked by reading, not compiled, not run on hardware
//
#include <hip/hip_runtime.h>

typedef __attribute__((ext_vector_type(16))) _Float16 v16h;
typedef __attribute__((ext_vector_type(8)))  float    v8f;
typedef __attribute__((ext_vector_type(4)))  float    v4f;

#define NROWS  65536
#define DDIM   64
#define KCB    1024
#define BB_    64
#define HH_    32
#define WW_    32
#define HW_    (HH_ * WW_)            // 1024
#define TILES  (NROWS / 16)           // 4096 16-row x-tiles
#define NTILES (KCB / 16)             // 64 codebook tiles

#if __has_builtin(__builtin_amdgcn_s_wait_asynccnt)
#define WAIT_ASYNC(n) __builtin_amdgcn_s_wait_asynccnt(n)
#else
#define WAIT_ASYNC(n) asm volatile("s_wait_asynccnt %0" :: "i"(n) : "memory")
#endif

// One async b128 (16B) copy global -> LDS per lane (ASYNCcnt-tracked).
__device__ __forceinline__ void async_copy_b128(unsigned lds_byte_off, const void* gsrc) {
    asm volatile("global_load_async_to_lds_b128 %0, %1, off"
                 :: "v"(lds_byte_off), "v"(gsrc) : "memory");
}

// ---------------------------------------------------------------------------
// Kernel 0: split codebook into f16 hi/lo and compute exact f32 row norms.
// ---------------------------------------------------------------------------
__global__ void prep_kernel(const float* __restrict__ w,
                            _Float16* __restrict__ w_hi,
                            _Float16* __restrict__ w_lo,
                            float* __restrict__ wsq) {
    __shared__ float red[DDIM];
    const int k = blockIdx.x;
    const int d = threadIdx.x;
    const float v = w[k * DDIM + d];
    const _Float16 hi = (_Float16)v;
    w_hi[k * DDIM + d] = hi;
    w_lo[k * DDIM + d] = (_Float16)(v - (float)hi);
    red[d] = v * v;
    __syncthreads();
    for (int s = DDIM / 2; s > 0; s >>= 1) {
        if (d < s) red[d] += red[d + s];
        __syncthreads();
    }
    if (d == 0) wsq[k] = red[0];
}

// ---------------------------------------------------------------------------
// Kernel 1: per-row argmin of ||w_n||^2 - 2 x.w_n via hi/lo split f16 WMMA.
// 8 waves/block; each wave owns TWO 16-row x-tiles. Codebook tiles are
// async-copied into a triple-buffered LDS stage shared by all 8 waves.
// ---------------------------------------------------------------------------
__global__ __launch_bounds__(256)
void argmin_kernel(const float* __restrict__ x,
                   const _Float16* __restrict__ w_hi,
                   const _Float16* __restrict__ w_lo,
                   const float* __restrict__ wsq,
                   int* __restrict__ out_idx) {
    __shared__ __align__(32) _Float16 sbuf[3][2 * 16 * DDIM]; // 3 x (hi 2KB | lo 2KB)
    __shared__ __align__(16) float    swsq[KCB];              // 4KB

    const int t    = threadIdx.x;
    const int lane = t & 31;
    const int wave = t >> 5;
    const int wt   = blockIdx.x * 8 + wave;     // global wave id (2 tiles each)
    const int lh   = lane & 15;
    const int sel  = lane >> 4;

    // --- staging role of this thread: one 16B segment of the 4KB tile ------
    const int shalf = t >> 7;                   // 0: hi half, 1: lo half
    const int srow  = (t >> 3) & 15;            // row within codebook tile
    const int sseg  = t & 7;                    // 8-half segment within row
    const _Float16* sgbase =
        (shalf ? w_lo : w_hi) + (size_t)srow * DDIM + sseg * 8;
    _Float16* sdst0 = &sbuf[0][shalf * (16 * DDIM) + srow * DDIM + sseg * 8];

    auto stage = [&](int nt, int b) {
        const _Float16* src = sgbase + (size_t)nt * 16 * DDIM;
        const unsigned dst =
            (unsigned)(uintptr_t)(sdst0 + (size_t)b * (2 * 16 * DDIM));
        async_copy_b128(dst, (const void*)src);
    };

    stage(0, 0);                                 // prime buffer 0

    // stage ||w||^2 into LDS (covered by the iter-0 barrier)
    *(v4f*)&swsq[t * 4] = *(const v4f*)&wsq[t * 4];

    // --- A fragments (hi/lo) for both M-tiles, K-chunks c=0,1 --------------
    // A 16x32 fragments as used here: lane-half holds K=e+16*sel (same order as B), M=lane&15.
    v16h a_hi[2][2], a_lo[2][2];
#pragma unroll
    for (int m = 0; m < 2; ++m) {
        const int row = wt * 32 + m * 16 + lh;
#pragma unroll
        for (int c = 0; c < 2; ++c) {
            const float* xp = x + (size_t)row * DDIM + 32 * c + 16 * sel;
            float xs[16];
            *(v4f*)&xs[0]  = *(const v4f*)(xp + 0);
            *(v4f*)&xs[4]  = *(const v4f*)(xp + 4);
            *(v4f*)&xs[8]  = *(const v4f*)(xp + 8);
            *(v4f*)&xs[12] = *(const v4f*)(xp + 12);
#pragma unroll
            for (int e = 0; e < 16; ++e) {
                const _Float16 h = (_Float16)xs[e];
                a_hi[m][c][e] = h;
                a_lo[m][c][e] = (_Float16)(xs[e] - (float)h);
            }
        }
    }

    float minval[2][8];
    int   minidx[2][8];
#pragma unroll
    for (int m = 0; m < 2; ++m)
#pragma unroll
        for (int r = 0; r < 8; ++r) { minval[m][r] = 3.0e38f; minidx[m][r] = 0; }

    for (int nt = 0; nt < NTILES; ++nt) {
        if (nt + 1 < NTILES) {                   // overlap copy of next tile
            stage(nt + 1, (nt + 1) % 3);
            WAIT_ASYNC(1);                       // tile nt landed (ours)
        } else {
            WAIT_ASYNC(0);
        }
        __syncthreads();                         // tile nt landed (all waves)

        // B 32x16 fragments from LDS: lane-half holds K=e+16*sel, col=lane&15
        const _Float16* bb = &sbuf[nt % 3][0];
        const v16h b_hi0 = *(const v16h*)(bb + lh * DDIM + 16 * sel);
        const v16h b_hi1 = *(const v16h*)(bb + lh * DDIM + 16 * sel + 32);
        const v16h b_lo0 = *(const v16h*)(bb + 16 * DDIM + lh * DDIM + 16 * sel);
        const v16h b_lo1 = *(const v16h*)(bb + 16 * DDIM + lh * DDIM + 16 * sel + 32);
        const int   ncol = nt * 16 + lh;
        const float wn2  = swsq[ncol];

#pragma unroll
        for (int m = 0; m < 2; ++m) {
            v8f acc = {};
            acc = __builtin_amdgcn_wmma_f32_16x16x32_f16(false, a_hi[m][0], false, b_hi0, (short)0, acc, false, false);
            acc = __builtin_amdgcn_wmma_f32_16x16x32_f16(false, a_hi[m][1], false, b_hi1, (short)0, acc, false, false);
            acc = __builtin_amdgcn_wmma_f32_16x16x32_f16(false, a_lo[m][0], false, b_hi0, (short)0, acc, false, false);
            acc = __builtin_amdgcn_wmma_f32_16x16x32_f16(false, a_lo[m][1], false, b_hi1, (short)0, acc, false, false);
            acc = __builtin_amdgcn_wmma_f32_16x16x32_f16(false, a_hi[m][0], false, b_lo0, (short)0, acc, false, false);
            acc = __builtin_amdgcn_wmma_f32_16x16x32_f16(false, a_hi[m][1], false, b_lo1, (short)0, acc, false, false);
#pragma unroll
            for (int r = 0; r < 8; ++r) {
                const float cand = wn2 - 2.0f * acc[r];
                if (cand < minval[m][r]) { minval[m][r] = cand; minidx[m][r] = ncol; }
            }
        }
    }

    // Reduce across the 16 columns in each lane-half; first-min tie-break.
#pragma unroll
    for (int off = 8; off >= 1; off >>= 1) {
#pragma unroll
        for (int m = 0; m < 2; ++m) {
#pragma unroll
            for (int r = 0; r < 8; ++r) {
                const float ov = __shfl_xor(minval[m][r], off, 16);
                const int   oi = __shfl_xor(minidx[m][r], off, 16);
                if (ov < minval[m][r] ||
                    (ov == minval[m][r] && oi < minidx[m][r])) {
                    minval[m][r] = ov; minidx[m][r] = oi;
                }
            }
        }
    }
    // lane 0 holds rows 0..7 of each tile, lane 16 rows 8..15.
    if (lh == 0) {
#pragma unroll
        for (int m = 0; m < 2; ++m)
#pragma unroll
            for (int r = 0; r < 8; ++r)
                out_idx[(wt * 2 + m) * 16 + 8 * sel + r] = minidx[m][r];
    }
}

// ---------------------------------------------------------------------------
// Kernel 2: gather (full f32 weights), transposed write-out, loss partials,
// and indices-as-float. 256 threads = 4 rows per block.
// ---------------------------------------------------------------------------
__global__ __launch_bounds__(256)
void gather_kernel(const float* __restrict__ x,
                   const float* __restrict__ w,
                   const int* __restrict__ idx,
                   float* __restrict__ out,
                   float* __restrict__ partial) {
    const int t = threadIdx.x;
    const int n = blockIdx.x * 4 + (t >> 6);
    const int d = t & 63;
    const int k = idx[n];
    const float q  = w[(size_t)k * DDIM + d];
    const float xv = x[(size_t)n * DDIM + d];
    const float diff = q - xv;

    // quantized reshape (B,H,W,D) -> transpose (B,D,H,W)
    const int b  = n >> 10;     // n / (H*W)
    const int hw = n & 1023;    // n % (H*W)
    out[1 + (size_t)b * (DDIM * HW_) + (size_t)d * HW_ + hw] = q;
    if (d == 0) out[1 + (size_t)BB_ * DDIM * HW_ + n] = (float)k;

    __shared__ float red[256];
    red[t] = diff * diff;
    __syncthreads();
    for (int s = 128; s > 0; s >>= 1) {
        if (t < s) red[t] += red[t + s];
        __syncthreads();
    }
    if (t == 0) partial[blockIdx.x] = red[0];
}

// ---------------------------------------------------------------------------
// Kernel 3: deterministic final loss reduction (fixed-order tree, no atomics).
// ---------------------------------------------------------------------------
__global__ void loss_kernel(const float* __restrict__ partial,
                            float* __restrict__ out) {
    __shared__ float red[256];
    const int t = threadIdx.x;
    float s = 0.0f;
    for (int i = t; i < NROWS / 4; i += 256) s += partial[i];
    red[t] = s;
    __syncthreads();
    for (int st = 128; st > 0; st >>= 1) {
        if (t < st) red[t] += red[t + st];
        __syncthreads();
    }
    if (t == 0) out[0] = red[0] * (2.0f / ((float)NROWS * (float)DDIM));
}

// ---------------------------------------------------------------------------
extern "C" void kernel_launch(void* const* d_in, const int* in_sizes, int n_in,
                              void* d_out, int out_size, void* d_ws, size_t ws_size,
                              hipStream_t stream) {
    const float* x = (const float*)d_in[0];   // [N, D] f32
    const float* w = (const float*)d_in[1];   // [K, D] f32
    float* out = (float*)d_out;               // [1 + B*D*H*W + N]

    // Workspace layout (all offsets 32B-aligned):
    char* p = (char*)d_ws;
    int*      ws_idx  = (int*)p;        p += (size_t)NROWS * sizeof(int);          // 256 KB
    float*    ws_wsq  = (float*)p;      p += (size_t)KCB * sizeof(float);          //   4 KB
    _Float16* ws_whi  = (_Float16*)p;   p += (size_t)KCB * DDIM * sizeof(_Float16);// 128 KB
    _Float16* ws_wlo  = (_Float16*)p;   p += (size_t)KCB * DDIM * sizeof(_Float16);// 128 KB
    float*    ws_part = (float*)p;      // NROWS/4 floats = 64 KB

    prep_kernel  <<<KCB,        DDIM, 0, stream>>>(w, ws_whi, ws_wlo, ws_wsq);
    argmin_kernel<<<TILES / 16, 256,  0, stream>>>(x, ws_whi, ws_wlo, ws_wsq, ws_idx);
    gather_kernel<<<NROWS / 4,  256,  0, stream>>>(x, w, ws_idx, out, ws_part);
    loss_kernel  <<<1,          256,  0, stream>>>(ws_part, out);
}
